// MutiheadRelativeAttention_3040836846167
// MI455X (gfx1250) — matrix operand order repaired and checked by reading, not compiled, not run on hardware
//
#include <hip/hip_runtime.h>
#include <hip/hip_bf16.h>
#include <cstdint>

typedef __bf16 bf16;
typedef __attribute__((ext_vector_type(16))) __bf16 v16bf;
typedef __attribute__((ext_vector_type(8)))  float  v8f;
typedef unsigned int u32x4 __attribute__((ext_vector_type(4)));
typedef int i32x4 __attribute__((ext_vector_type(4)));
typedef int i32x8 __attribute__((ext_vector_type(8)));

#define WMMA_BF16(a, b, c) \
  __builtin_amdgcn_wmma_f32_16x16x32_bf16(false, (a), false, (b), (short)0, (c), false, false)

#define REL_PAD 288  // 257 rel positions padded to a multiple of 32 (K dim)

// ---------------------------------------------------------------------------
// CDNA5 async / tensor data movement helpers
// ---------------------------------------------------------------------------

// Generic pointers to LDS carry the wave-relative LDS byte offset in the low
// 32 bits (see ISA 10.2).
__device__ inline unsigned lds_addr_of(const void* p) {
  return (unsigned)(unsigned long long)(uintptr_t)p;
}

// 16B global -> LDS async copy (per active lane). Tracked by ASYNCcnt.
__device__ inline void async_copy_b128(unsigned lds_off, const void* gptr) {
  unsigned long long ga = (unsigned long long)(uintptr_t)gptr;
  asm volatile("global_load_async_to_lds_b128 %0, %1, off"
               :: "v"(lds_off), "v"(ga) : "memory");
}

__device__ inline void async_wait0() {
  asm volatile("s_wait_asynccnt 0x0" ::: "memory");
}

// One 2D tensor-DMA tile load (rows x cols bf16, row stride in elements)
// issued per wave (TDM ignores EXEC; issue from one wave per workgroup).
__device__ inline void tdm_load_2d_bf16(unsigned lds_off, const void* gptr,
                                        unsigned tile_cols, unsigned tile_rows,
                                        unsigned row_stride_elems) {
  unsigned long long ga = (unsigned long long)(uintptr_t)gptr;
  u32x4 g0;
  g0[0] = 1u;                                 // count=1, user descriptor
  g0[1] = lds_off;                            // lds_addr
  g0[2] = (unsigned)ga;                       // global_addr[31:0]
  g0[3] = (unsigned)(ga >> 32) | (2u << 30);  // global_addr[56:32] | type=2
  i32x8 g1;
  g1[0] = (int)(1u << 16);                    // data_size=1 (2 bytes)
  g1[1] = (int)((row_stride_elems & 0xffffu) << 16);        // tensor_dim0 lo16
  g1[2] = (int)((row_stride_elems >> 16) & 0xffffu);        // tensor_dim0 hi16
  g1[3] = (int)(0x10u | (tile_cols << 16));   // tensor_dim1=1<<20, tile_dim0
  g1[4] = (int)tile_rows;                     // tile_dim1 (tile_dim2=0)
  g1[5] = (int)row_stride_elems;              // tensor_dim0_stride[31:0]
  g1[6] = 0;                                  // stride hi, dim1_stride lo
  g1[7] = 0;
  i32x4 z4 = {};
#if defined(__clang_major__) && (__clang_major__ >= 23)
  i32x8 z8 = {};
  __builtin_amdgcn_tensor_load_to_lds(g0, g1, z4, z4, z8, 0);
#else
  __builtin_amdgcn_tensor_load_to_lds(g0, g1, z4, z4, 0);
#endif
}

__device__ inline void tensor_wait0() {
  __builtin_amdgcn_s_wait_tensorcnt(0);
}

// ---------------------------------------------------------------------------
// WMMA fragment loaders (CDNA5 wave32 VGPR layouts, 16x16x32 bf16)
// ---------------------------------------------------------------------------

// A operand: 16x32 (MxK), storage row-major p[m][k] with given stride.
__device__ inline v16bf frag_a_lds(const bf16* p, int stride) {
  const int lane = threadIdx.x & 31;
  const int row  = lane & 15;
  const int kb   = (lane >> 4) << 4;  // 0 or 16
  const bf16* r = p + row * stride;
  v16bf a;
#pragma unroll
  for (int e = 0; e < 16; ++e) {
    int k = kb + e;
    a[e] = r[k];
  }
  return a;
}

// B operand: 32x16 (KxN), storage is B^T row-major: p[n][k].
__device__ inline v16bf frag_b_lds(const bf16* p, int stride) {
  const int lane = threadIdx.x & 31;
  const int n  = lane & 15;
  const int kb = (lane >> 4) << 4;  // 0 or 16
  const bf16* r = p + n * stride + kb;
  v16bf b;
#pragma unroll
  for (int e = 0; e < 16; ++e) b[e] = r[e];
  return b;
}

// B operand from K-major storage p[k][n] (e.g. a V tile stored row-major).
__device__ inline v16bf frag_b_lds_kmaj(const bf16* p, int stride) {
  const int lane = threadIdx.x & 31;
  const int n  = lane & 15;
  const int kb = (lane >> 4) << 4;
  v16bf b;
#pragma unroll
  for (int e = 0; e < 16; ++e) b[e] = p[(kb + e) * stride + n];
  return b;
}

// ---------------------------------------------------------------------------
// Generic bf16 WMMA GEMM: C[M,N] = A[M,K] * B[N,K]^T + bias
// Double-buffered LDS staging via async global->LDS copies overlapping WMMA.
// ---------------------------------------------------------------------------
__global__ __launch_bounds__(128)
void gemm_xwT(const bf16* __restrict__ A16, const float* __restrict__ A32,
              const bf16* __restrict__ Bw, const float* __restrict__ bias,
              float* __restrict__ C32, bf16* __restrict__ C16,
              int M, int N, int K)
{
  __shared__ __align__(16) bf16 At[2][64 * 32];
  __shared__ __align__(16) bf16 Bt[2][64 * 32];

  const int tid  = threadIdx.x;
  const int wv   = tid >> 5;
  const int lane = tid & 31;
  const int nl     = lane & 15;
  const int mf_hi  = (lane >> 4) << 3;  // C-frag row offset (0 or 8)
  const int mbase = blockIdx.y * 64;
  const int nbase = blockIdx.x * 64;

  v8f acc[4] = {};

  auto stage = [&](int buf, int k0) {
    if (A16) {
#pragma unroll
      for (int t = 0; t < 2; ++t) {
        int chunk = tid + t * 128;      // 256 chunks of 8 bf16
        int r = chunk >> 2;
        int c = (chunk & 3) << 3;
        async_copy_b128(lds_addr_of(&At[buf][r * 32 + c]),
                        &A16[(size_t)(mbase + r) * K + k0 + c]);
      }
    } else {
      for (int i = tid; i < 64 * 32; i += 128) {
        int r = i >> 5, c = i & 31;
        At[buf][i] = (bf16)A32[(size_t)(mbase + r) * K + k0 + c];
      }
    }
#pragma unroll
    for (int t = 0; t < 2; ++t) {
      int chunk = tid + t * 128;
      int r = chunk >> 2;
      int c = (chunk & 3) << 3;
      async_copy_b128(lds_addr_of(&Bt[buf][r * 32 + c]),
                      &Bw[(size_t)(nbase + r) * K + k0 + c]);
    }
  };

  stage(0, 0);
  int cur = 0;
  for (int k0 = 0; k0 < K; k0 += 32) {
    async_wait0();
    __syncthreads();
    if (k0 + 32 < K) stage(cur ^ 1, k0 + 32);  // DMA next slice during WMMA

    v16bf a = frag_a_lds(&At[cur][wv * 16 * 32], 32);
#pragma unroll
    for (int j = 0; j < 4; ++j) {
      v16bf b = frag_b_lds(&Bt[cur][j * 16 * 32], 32);
      acc[j] = WMMA_BF16(a, b, acc[j]);
    }
    cur ^= 1;
  }

#pragma unroll
  for (int j = 0; j < 4; ++j) {
    int col = nbase + j * 16 + nl;
    float bv = bias ? bias[col] : 0.f;
#pragma unroll
    for (int r = 0; r < 8; ++r) {
      int row = mbase + wv * 16 + r + mf_hi;
      float v = acc[j][r] + bv;
      if (C32) C32[(size_t)row * N + col] = v;
      if (C16) C16[(size_t)row * N + col] = (bf16)v;
    }
  }
}

// ---------------------------------------------------------------------------
// Fused relative attention, one (b0,h0) block, 32 q-rows per WG, 2 waves.
// Two-pass flash softmax; rel-K bias via WMMA-precomputed S2 table in LDS;
// rel-V handled by scattering final attn into Prel[q, 288] buckets (global).
// K tiles staged with async-to-LDS; V tiles staged with the Tensor Data Mover.
// ---------------------------------------------------------------------------
__global__ __launch_bounds__(64)
void attn_kernel(const bf16* __restrict__ Qbf, const bf16* __restrict__ Kbf,
                 const bf16* __restrict__ Vbf, const bf16* __restrict__ Rkp,
                 float* __restrict__ prelg, float* __restrict__ buf1)
{
  __shared__ __align__(16) bf16 s2[32][REL_PAD];
  __shared__ __align__(16) bf16 qt[32][64];
  __shared__ __align__(16) bf16 kt[64][64];
  __shared__ __align__(16) bf16 vt[64][64];
  __shared__ __align__(16) bf16 pt[2][16][64];

  const int tid  = threadIdx.x;
  const int wv   = tid >> 5;
  const int lane = tid & 31;
  const int nl     = lane & 15;
  const int mf_hi  = (lane >> 4) << 3;

  const int q0 = blockIdx.x * 32;
  const int m0 = blockIdx.y;          // attention block index = b0*16 + h0
  const int b0 = m0 >> 4, h0 = m0 & 15;
  // reference quirk: score_2 for (b0,h0) uses q_head of m' = h0*B + b0
  const int mq = h0 * 4 + b0;
  const int bq = mq >> 4, hq = mq & 15;

  // Zero this block's Prel rows (owned exclusively; deterministic init).
  float* prow = prelg + ((size_t)m0 * 1024 + q0) * REL_PAD;
  for (int i = tid; i < 32 * REL_PAD; i += 64) prow[i] = 0.f;

  // Stage own Q tile and permuted-Q (into kt as temp) via async copies.
  for (int ch = tid; ch < 32 * 8; ch += 64) {
    int r = ch >> 3, c = (ch & 7) << 3;
    async_copy_b128(lds_addr_of(&qt[r][c]),
                    &Qbf[(size_t)(b0 * 1024 + q0 + r) * 1024 + h0 * 64 + c]);
    async_copy_b128(lds_addr_of(&kt[r][c]),
                    &Qbf[(size_t)(bq * 1024 + q0 + r) * 1024 + hq * 64 + c]);
  }
  async_wait0();
  __syncthreads();

  // S2[q, r] = Qperm[q,:] . rel_k_pad[r,:]   (32 x 288 via WMMA)
  {
    v16bf a0 = frag_a_lds(&kt[wv * 16][0], 64);
    v16bf a1 = frag_a_lds(&kt[wv * 16][32], 64);
    for (int j = 0; j < REL_PAD / 16; ++j) {
      v8f acc = {};
      acc = WMMA_BF16(a0, frag_b_lds(&Rkp[(j * 16) * 64 + 0], 64), acc);
      acc = WMMA_BF16(a1, frag_b_lds(&Rkp[(j * 16) * 64 + 32], 64), acc);
#pragma unroll
      for (int r = 0; r < 8; ++r)
        s2[wv * 16 + r + mf_hi][j * 16 + nl] = (bf16)acc[r];
    }
  }
  __syncthreads();

  const float scale = 0.125f;  // 1/sqrt(D=64)
  float mrow[8], srow[8];
#pragma unroll
  for (int r = 0; r < 8; ++r) { mrow[r] = -1e30f; srow[r] = 0.f; }

  // ---- Pass 1: softmax statistics ----
  for (int ks = 0; ks < 1024; ks += 64) {
    for (int ch = tid; ch < 64 * 8; ch += 64) {
      int r = ch >> 3, c = (ch & 7) << 3;
      async_copy_b128(lds_addr_of(&kt[r][c]),
                      &Kbf[(size_t)(b0 * 1024 + ks + r) * 1024 + h0 * 64 + c]);
    }
    async_wait0();
    __syncthreads();

    v16bf a0 = frag_a_lds(&qt[wv * 16][0], 64);
    v16bf a1 = frag_a_lds(&qt[wv * 16][32], 64);
    float sv[4][8];
#pragma unroll
    for (int j = 0; j < 4; ++j) {
      v8f acc = {};
      acc = WMMA_BF16(a0, frag_b_lds(&kt[j * 16][0], 64), acc);
      acc = WMMA_BF16(a1, frag_b_lds(&kt[j * 16][32], 64), acc);
      int k = ks + j * 16 + nl;
#pragma unroll
      for (int r = 0; r < 8; ++r) {
        int qloc = wv * 16 + r + mf_hi;
        int q = q0 + qloc;
        int d = q - k; d = d < -128 ? -128 : (d > 128 ? 128 : d);
        sv[j][r] = (acc[r] + (float)s2[qloc][d + 128]) * scale;
      }
    }
#pragma unroll
    for (int r = 0; r < 8; ++r) {
      float tm = sv[0][r];
#pragma unroll
      for (int j = 1; j < 4; ++j) tm = fmaxf(tm, sv[j][r]);
#pragma unroll
      for (int m = 8; m >= 1; m >>= 1) tm = fmaxf(tm, __shfl_xor(tm, m, 16));
      float mnew = fmaxf(mrow[r], tm);
      float es = 0.f;
#pragma unroll
      for (int j = 0; j < 4; ++j) es += __expf(sv[j][r] - mnew);
#pragma unroll
      for (int m = 8; m >= 1; m >>= 1) es += __shfl_xor(es, m, 16);
      srow[r] = srow[r] * __expf(mrow[r] - mnew) + es;
      mrow[r] = mnew;
    }
    __syncthreads();
  }

  float rinv[8];
#pragma unroll
  for (int r = 0; r < 8; ++r) rinv[r] = 1.f / srow[r];

  // ---- Pass 2: exact P, accumulate O1 = P·V, scatter P into Prel buckets ----
  v8f oacc[4] = {};
  for (int ks = 0; ks < 1024; ks += 64) {
    // V strip via Tensor Data Mover (one descriptor per workgroup, wave 0).
    if (wv == 0) {
      tdm_load_2d_bf16(lds_addr_of(&vt[0][0]),
                       &Vbf[(size_t)(b0 * 1024 + ks) * 1024 + h0 * 64],
                       64, 64, 1024);
    }
    // K strip via async global->LDS copies (all lanes).
    for (int ch = tid; ch < 64 * 8; ch += 64) {
      int r = ch >> 3, c = (ch & 7) << 3;
      async_copy_b128(lds_addr_of(&kt[r][c]),
                      &Kbf[(size_t)(b0 * 1024 + ks + r) * 1024 + h0 * 64 + c]);
    }
    async_wait0();
    tensor_wait0();
    __syncthreads();

    v16bf a0 = frag_a_lds(&qt[wv * 16][0], 64);
    v16bf a1 = frag_a_lds(&qt[wv * 16][32], 64);
#pragma unroll
    for (int j = 0; j < 4; ++j) {
      v8f acc = {};
      acc = WMMA_BF16(a0, frag_b_lds(&kt[j * 16][0], 64), acc);
      acc = WMMA_BF16(a1, frag_b_lds(&kt[j * 16][32], 64), acc);
      int k = ks + j * 16 + nl;
#pragma unroll
      for (int r = 0; r < 8; ++r) {
        int qloc = wv * 16 + r + mf_hi;
        int q = q0 + qloc;
        int d = q - k; d = d < -128 ? -128 : (d > 128 ? 128 : d);
        float val = (acc[r] + (float)s2[qloc][d + 128]) * scale;
        float p = __expf(val - mrow[r]) * rinv[r];
        pt[wv][r + mf_hi][j * 16 + nl] = (bf16)p;
        atomicAdd(&prow[(size_t)qloc * REL_PAD + (d + 128)], p);
      }
    }
    // O1 += P · V  (P staged via wave-local LDS; DS ops are in-order per wave)
    v16bf pa0 = frag_a_lds(&pt[wv][0][0], 64);
    v16bf pa1 = frag_a_lds(&pt[wv][0][32], 64);
#pragma unroll
    for (int dj = 0; dj < 4; ++dj) {
      oacc[dj] = WMMA_BF16(pa0, frag_b_lds_kmaj(&vt[0][dj * 16], 64), oacc[dj]);
      oacc[dj] = WMMA_BF16(pa1, frag_b_lds_kmaj(&vt[32][dj * 16], 64), oacc[dj]);
    }
    __syncthreads();
  }

  // Store weight_1 into [B,T,E] layout (e = h0*64 + d)
#pragma unroll
  for (int dj = 0; dj < 4; ++dj) {
#pragma unroll
    for (int r = 0; r < 8; ++r) {
      int q = q0 + wv * 16 + r + mf_hi;
      int col = h0 * 64 + dj * 16 + nl;
      buf1[(size_t)(b0 * 1024 + q) * 1024 + col] = oacc[dj][r];
    }
  }
}

// ---------------------------------------------------------------------------
// Small helper kernels
// ---------------------------------------------------------------------------
__global__ void cvt_bf16_kernel(const float* __restrict__ in, bf16* __restrict__ out, int n) {
  int i = blockIdx.x * blockDim.x + threadIdx.x;
  if (i < n) out[i] = (bf16)in[i];
}

__global__ void pad_rk_kernel(const float* __restrict__ rel_k, bf16* __restrict__ rkp) {
  int i = blockIdx.x * blockDim.x + threadIdx.x;  // [288][64]
  if (i >= REL_PAD * 64) return;
  int r = i >> 6, d = i & 63;
  rkp[i] = (bf16)(r < 257 ? rel_k[r * 64 + d] : 0.f);
}

__global__ void pad_rvT_kernel(const float* __restrict__ rel_v, bf16* __restrict__ rvT) {
  int i = blockIdx.x * blockDim.x + threadIdx.x;  // [64][288] = rel_v^T padded
  if (i >= 64 * REL_PAD) return;
  int d = i / REL_PAD, r = i % REL_PAD;
  rvT[i] = (bf16)(r < 257 ? rel_v[r * 64 + d] : 0.f);
}

// attnA[b,t,e] = bf16( weight_1[b,t,e] + weight_2[m0=h*4+b][t][d] ), e=h*64+d
__global__ void combine_kernel(const float* __restrict__ buf1, const float* __restrict__ buf2,
                               bf16* __restrict__ attnA) {
  size_t i = (size_t)blockIdx.x * blockDim.x + threadIdx.x;  // over 4*1024*1024
  int e = (int)(i & 1023);
  size_t bt = i >> 10;
  int t = (int)(bt & 1023);
  int b = (int)(bt >> 10);
  int h = e >> 6, d = e & 63;
  int m0 = h * 4 + b;  // reference (b,h)-packing quirk for weight_2
  float v = buf1[i] + buf2[((size_t)m0 * 1024 + t) * 64 + d];
  attnA[i] = (bf16)v;
}

// ---------------------------------------------------------------------------
// Host launcher
// ---------------------------------------------------------------------------
extern "C" void kernel_launch(void* const* d_in, const int* in_sizes, int n_in,
                              void* d_out, int out_size, void* d_ws, size_t ws_size,
                              hipStream_t stream)
{
  (void)in_sizes; (void)n_in; (void)out_size; (void)ws_size;
  const float* hidden = (const float*)d_in[0];
  const float* Wq  = (const float*)d_in[1];
  const float* bqv = (const float*)d_in[2];
  const float* Wk  = (const float*)d_in[3];
  const float* bkv = (const float*)d_in[4];
  const float* Wv  = (const float*)d_in[5];
  const float* bvv = (const float*)d_in[6];
  const float* Wo  = (const float*)d_in[7];
  const float* bov = (const float*)d_in[8];
  const float* rel_k = (const float*)d_in[9];
  const float* rel_v = (const float*)d_in[10];

  const int T = 1024, E = 1024;
  const size_t MT = 4096;  // B*T

  char* ws = (char*)d_ws;
  size_t off = 0;
  auto alloc = [&](size_t bytes) -> void* {
    void* p = ws + off;
    off += (bytes + 255) & ~(size_t)255;
    return p;
  };

  bf16* hid_bf = (bf16*)alloc(MT * E * 2);
  bf16* wq_bf  = (bf16*)alloc((size_t)E * E * 2);
  bf16* wk_bf  = (bf16*)alloc((size_t)E * E * 2);
  bf16* wv_bf  = (bf16*)alloc((size_t)E * E * 2);
  bf16* wo_bf  = (bf16*)alloc((size_t)E * E * 2);
  bf16* q_bf   = (bf16*)alloc(MT * E * 2);
  bf16* k_bf   = (bf16*)alloc(MT * E * 2);
  bf16* v_bf   = (bf16*)alloc(MT * E * 2);
  bf16* rkp    = (bf16*)alloc((size_t)REL_PAD * 64 * 2);
  bf16* rvT    = (bf16*)alloc((size_t)64 * REL_PAD * 2);
  float* prelg = (float*)alloc((size_t)64 * 1024 * REL_PAD * 4);
  float* buf1  = (float*)alloc(MT * E * 4);
  float* buf2  = (float*)alloc((size_t)64 * 1024 * 64 * 4);
  bf16* attnA  = (bf16*)alloc(MT * E * 2);

  int n = (int)(MT * E);
  cvt_bf16_kernel<<<(n + 255) / 256, 256, 0, stream>>>(hidden, hid_bf, n);
  int nw = E * E;
  cvt_bf16_kernel<<<(nw + 255) / 256, 256, 0, stream>>>(Wq, wq_bf, nw);
  cvt_bf16_kernel<<<(nw + 255) / 256, 256, 0, stream>>>(Wk, wk_bf, nw);
  cvt_bf16_kernel<<<(nw + 255) / 256, 256, 0, stream>>>(Wv, wv_bf, nw);
  cvt_bf16_kernel<<<(nw + 255) / 256, 256, 0, stream>>>(Wo, wo_bf, nw);
  pad_rk_kernel<<<(REL_PAD * 64 + 255) / 256, 256, 0, stream>>>(rel_k, rkp);
  pad_rvT_kernel<<<(64 * REL_PAD + 255) / 256, 256, 0, stream>>>(rel_v, rvT);

  dim3 gProj(E / 64, (int)(MT / 64));  // (16, 64)
  gemm_xwT<<<gProj, 128, 0, stream>>>(hid_bf, nullptr, wq_bf, bqv, nullptr, q_bf, (int)MT, E, E);
  gemm_xwT<<<gProj, 128, 0, stream>>>(hid_bf, nullptr, wk_bf, bkv, nullptr, k_bf, (int)MT, E, E);
  gemm_xwT<<<gProj, 128, 0, stream>>>(hid_bf, nullptr, wv_bf, bvv, nullptr, v_bf, (int)MT, E, E);

  attn_kernel<<<dim3(T / 32, 64), 64, 0, stream>>>(q_bf, k_bf, v_bf, rkp, prelg, buf1);

  // weight_2 = Prel[64*1024, 288] · rel_v_pad  ->  buf2[64*1024, 64]
  gemm_xwT<<<dim3(1, 64 * 1024 / 64), 128, 0, stream>>>(
      nullptr, prelg, rvT, nullptr, buf2, nullptr, 64 * 1024, 64, REL_PAD);

  combine_kernel<<<(int)((MT * E + 255) / 256), 256, 0, stream>>>(buf1, buf2, attnA);

  gemm_xwT<<<gProj, 128, 0, stream>>>(attnA, nullptr, wo_bf, bov, (float*)d_out, nullptr,
                                      (int)MT, E, E);
}
